// ContrastiveLoss_37606733644514
// MI455X (gfx1250) — matrix operand order repaired and checked by reading, not compiled, not run on hardware
//
#include <hip/hip_runtime.h>
#include <hip/hip_bf16.h>
#include <math.h>
#include <stdint.h>

typedef __attribute__((ext_vector_type(16))) _Float16 v16h;
typedef __attribute__((ext_vector_type(8)))  _Float16 v8h;
typedef __attribute__((ext_vector_type(4)))  _Float16 v4h;
typedef __attribute__((ext_vector_type(8)))  float    v8f;
typedef __attribute__((ext_vector_type(4)))  float    v4f;
typedef __attribute__((ext_vector_type(4)))  int      v4i;
typedef __attribute__((address_space(1)))    v4i      g_v4i;   // global int4
typedef __attribute__((address_space(3)))    v4i      l_v4i;   // LDS int4

#define BATCH   4096
#define TWO_N   8192
#define DIM     128
#define N_TILES 512            // TWO_N / 16
#define POS_SHIFT 256          // BATCH / 16
#define CSPLIT  8              // column-range splits (blockIdx.y)
#define RG_WAVES 8             // row tiles (waves) per block
#define CT_PER_BLOCK (N_TILES / CSPLIT)   // 64 column tiles per block
#define LDS_PITCH 136          // halves per staged row (pad: conflict-free)
// exp(x/TEMP) == exp2(x * LOG2E/TEMP);  TEMP = 0.5
#define SIM_SCALE 2.8853900817779268f   // 2 * log2(e)
#define LN2       0.6931471805599453f

#if defined(__has_builtin)
#if __has_builtin(__builtin_amdgcn_global_load_async_to_lds_b128) && \
    __has_builtin(__builtin_amdgcn_s_wait_asynccnt)
#define USE_ASYNC_LDS 1
#endif
#if __has_builtin(__builtin_amdgcn_sched_barrier)
#define SCHED_FENCE() __builtin_amdgcn_sched_barrier(0)
#endif
#endif
#ifndef SCHED_FENCE
#define SCHED_FENCE()
#endif

// ---------------------------------------------------------------------------
// Kernel 1: row-normalize z = [z_i ; z_j]  ->  f16 matrix zn [8192 x 128]
// ---------------------------------------------------------------------------
__global__ void ntxent_normalize(const float* __restrict__ zi,
                                 const float* __restrict__ zj,
                                 _Float16* __restrict__ zn) {
  const int wave = threadIdx.x >> 5;          // 8 waves / block
  const int lane = threadIdx.x & 31;
  const int row  = blockIdx.x * 8 + wave;     // grid = 1024 blocks
  const float* src = (row < BATCH) ? (zi + row * DIM)
                                   : (zj + (row - BATCH) * DIM);
  v4f v = *(const v4f*)(src + lane * 4);
  float ss = v.x * v.x + v.y * v.y + v.z * v.z + v.w * v.w;
  #pragma unroll
  for (int m = 1; m < 32; m <<= 1)
    ss += __shfl_xor(ss, m, 32);
  const float inv = rsqrtf(ss);
  v4h h;
  h.x = (_Float16)(v.x * inv);
  h.y = (_Float16)(v.y * inv);
  h.z = (_Float16)(v.z * inv);
  h.w = (_Float16)(v.w * inv);
  *(v4h*)(zn + row * DIM + lane * 4) = h;
}

// stage one 16-byte chunk of a column tile into LDS
__device__ __forceinline__ void stage_chunk(const _Float16* __restrict__ g,
                                            _Float16* l) {
#ifdef USE_ASYNC_LDS
  __builtin_amdgcn_global_load_async_to_lds_b128(
      (g_v4i*)(uintptr_t)g, (l_v4i*)(uint32_t)(uintptr_t)l, 0, 0);
#else
  *(v8h*)l = *(const v8h*)g;
#endif
}

__device__ __forceinline__ void async_fence() {
#ifdef USE_ASYNC_LDS
  __builtin_amdgcn_s_wait_asynccnt(0);
#endif
}

// ---------------------------------------------------------------------------
// Kernel 2: fused sim-GEMM + masked streaming logsumexp.
// Block = 8 waves, each owning one 16-row tile of a 128-row group; all waves
// share each staged 16x128 f16 column tile via LDS (async DMA, loaded ONCE
// per block). Column range split CSPLIT ways across blockIdx.y; per-row
// partial sums go to workspace for the final combine.
// ---------------------------------------------------------------------------
__global__ void ntxent_loss(const _Float16* __restrict__ zn,
                            float* __restrict__ rowsum,
                            float* __restrict__ pospart) {
  __shared__ _Float16 sbuf[2][16 * LDS_PITCH];

  const int wid  = threadIdx.x >> 5;                 // 0..7
  const int lane = threadIdx.x & 31;
  const int rg   = blockIdx.x;                       // row group 0..63
  const int cs   = blockIdx.y;                       // column split 0..7
  const int tile = rg * RG_WAVES + wid;              // row tile 0..511
  const int ct0  = cs * CT_PER_BLOCK;
  const int r16  = lane & 15;

  // 0/1 mask for the diagonal C/D slot owned by this lane (branchless fixups)
  const int diagv = (lane < 8) ? lane : ((lane >= 24) ? (lane - 24) : -1);
  float dmask[8];
  #pragma unroll
  for (int v = 0; v < 8; ++v) dmask[v] = (v == diagv) ? 1.0f : 0.0f;
  const int postile = (tile + POS_SHIFT) & (N_TILES - 1);

  // ---- load the 4 A fragments for this wave's row tile (registers) ----
  const _Float16* arow = zn + (tile * 16 + r16) * DIM;
  const int aoff = (lane < 16) ? 0 : 8;
  v16h a0, a1, a2, a3;
  #pragma unroll
  for (int kk = 0; kk < 4; ++kk) {
    const _Float16* p = arow + kk * 32 + aoff;
    v8h lo = *(const v8h*)(p);
    v8h hi = *(const v8h*)(p + 16);
    v16h a;
    #pragma unroll
    for (int i = 0; i < 8; ++i) { a[i] = lo[i]; a[i + 8] = hi[i]; }
    if (kk == 0) a0 = a; else if (kk == 1) a1 = a;
    else if (kk == 2) a2 = a; else a3 = a;
  }

  float s[8];
  #pragma unroll
  for (int v = 0; v < 8; ++v) s[v] = 0.0f;
  float possum = 0.0f;                               // in log2 units

  // staging role of this thread: one b128 chunk of the 16x128 tile
  const int srow = threadIdx.x >> 4;                 // 0..15
  const int schk = threadIdx.x & 15;                 // 0..15
  const _Float16* gstage = zn + srow * DIM + schk * 8;     // + ct*16*DIM
  _Float16* l0 = &sbuf[0][srow * LDS_PITCH + schk * 8];
  _Float16* l1 = &sbuf[1][srow * LDS_PITCH + schk * 8];

  const int boff = (lane < 16) ? 0 : 8;

  auto compute = [&](const _Float16* buf, int ct) {
    const _Float16* base = buf + r16 * LDS_PITCH + boff;
    v16h b0, b1, b2, b3;
    #pragma unroll
    for (int kk = 0; kk < 4; ++kk) {
      v8h lo = *(const v8h*)(base + kk * 32);
      v8h hi = *(const v8h*)(base + kk * 32 + 16);
      v16h b;
      #pragma unroll
      for (int i = 0; i < 8; ++i) { b[i] = lo[i]; b[i + 8] = hi[i]; }
      if (kk == 0) b0 = b; else if (kk == 1) b1 = b;
      else if (kk == 2) b2 = b; else b3 = b;
    }
    SCHED_FENCE();   // issue all 8 ds_load_b128 before the WMMA chain
    v8f c = {};
    c = __builtin_amdgcn_wmma_f32_16x16x32_f16(false, a0, false, b0, (short)0, c, false, false);
    c = __builtin_amdgcn_wmma_f32_16x16x32_f16(false, a1, false, b1, (short)0, c, false, false);
    c = __builtin_amdgcn_wmma_f32_16x16x32_f16(false, a2, false, b2, (short)0, c, false, false);
    c = __builtin_amdgcn_wmma_f32_16x16x32_f16(false, a3, false, b3, (short)0, c, false, false);

    float x[8], e[8];
    #pragma unroll
    for (int v = 0; v < 8; ++v) {
      x[v] = c[v] * SIM_SCALE;                       // log2-domain logits
      e[v] = __builtin_amdgcn_exp2f(x[v]);           // bare v_exp_f32
      s[v] += e[v];
    }
    if (ct == tile) {                                // remove self-similarity
      #pragma unroll
      for (int v = 0; v < 8; ++v) s[v] = fmaf(-dmask[v], e[v], s[v]);
    }
    if (ct == postile) {                             // capture positive logit
      #pragma unroll
      for (int v = 0; v < 8; ++v) possum = fmaf(dmask[v], x[v], possum);
    }
  };

  // ---- software-pipelined main loop over this block's column tiles ----
  stage_chunk(gstage + (size_t)ct0 * 16 * DIM, l0);
  async_fence();
  __syncthreads();

  for (int k = 0; k < CT_PER_BLOCK; k += 2) {
    const int ct = ct0 + k;
    stage_chunk(gstage + (size_t)(ct + 1) * 16 * DIM, l1);   // DMA overlaps
    compute(sbuf[0], ct);
    async_fence();
    __syncthreads();

    if (k + 2 < CT_PER_BLOCK)
      stage_chunk(gstage + (size_t)(ct + 2) * 16 * DIM, l0);
    compute(sbuf[1], ct + 1);
    async_fence();
    __syncthreads();
  }

  // ---- intra-wave: butterfly row sums across 16 lanes of each half ----
  #pragma unroll
  for (int v = 0; v < 8; ++v) {
    #pragma unroll
    for (int m = 1; m < 16; m <<= 1)
      s[v] += __shfl_xor(s[v], m, 32);
  }
  #pragma unroll
  for (int m = 1; m < 32; m <<= 1)
    possum += __shfl_xor(possum, m, 32);

  // publish per-row partial sums (column-split cs) + positive partials
  v8f vs;
  #pragma unroll
  for (int v = 0; v < 8; ++v) vs[v] = s[v];
  float* rbase = rowsum + (size_t)cs * TWO_N + tile * 16;
  if (lane == 0) {
    *(v8f*)rbase = vs;                               // rows 0..7 of tile
    pospart[cs * N_TILES + tile] = possum;
  }
  if (lane == 16)
    *(v8f*)(rbase + 8) = vs;                         // rows 8..15 of tile
}

// ---------------------------------------------------------------------------
// Kernel 3: combine column-split partials per row, deterministic reduction
// ---------------------------------------------------------------------------
__global__ void ntxent_reduce(const float* __restrict__ rowsum,
                              const float* __restrict__ pospart,
                              float* __restrict__ out) {
  __shared__ float sm[1024];
  const int t = threadIdx.x;                         // 1024 threads, 1 block
  float acc = 0.0f;
  #pragma unroll
  for (int i = 0; i < TWO_N / 1024; ++i) {           // 8 rows per thread
    const int row = t + i * 1024;
    float tot = 0.0f;
    #pragma unroll
    for (int c = 0; c < CSPLIT; ++c)
      tot += rowsum[c * TWO_N + row];
    acc += __builtin_amdgcn_logf(tot);               // bare v_log_f32 (log2)
  }
  #pragma unroll
  for (int c = 0; c < (CSPLIT * N_TILES) / 1024; ++c)   // 4096 pos partials
    acc -= pospart[t + c * 1024];
  sm[t] = acc;
  __syncthreads();
  #pragma unroll
  for (int stride = 512; stride > 0; stride >>= 1) {
    if (t < stride) sm[t] += sm[t + stride];
    __syncthreads();
  }
  if (t == 0) out[0] = sm[0] * (LN2 / (float)TWO_N);
}

// ---------------------------------------------------------------------------
extern "C" void kernel_launch(void* const* d_in, const int* in_sizes, int n_in,
                              void* d_out, int out_size, void* d_ws, size_t ws_size,
                              hipStream_t stream) {
  const float* zi = (const float*)d_in[0];
  const float* zj = (const float*)d_in[1];
  char* ws = (char*)d_ws;
  _Float16* zn   = (_Float16*)ws;                        // 2 MB
  float* rowsum  = (float*)(ws + (size_t)TWO_N * DIM * 2);         // 256 KB
  float* pospart = (float*)(ws + (size_t)TWO_N * DIM * 2 +
                            (size_t)CSPLIT * TWO_N * 4);           // 16 KB
  float* out = (float*)d_out;

  ntxent_normalize<<<TWO_N / 8, 256, 0, stream>>>(zi, zj, zn);
  ntxent_loss<<<dim3(N_TILES / RG_WAVES, CSPLIT), RG_WAVES * 32, 0, stream>>>(
      zn, rowsum, pospart);
  ntxent_reduce<<<1, 1024, 0, stream>>>(rowsum, pospart, out);
}
